// DgcnNetwork_28501402976372
// MI455X (gfx1250) — matrix operand order repaired and checked by reading, not compiled, not run on hardware
//
#include <hip/hip_runtime.h>
#include <math.h>
#include <stdint.h>

#define BN_EPS 1e-5f

// ---------------------------------------------------------------------------
// CDNA5 WMMA types
// ---------------------------------------------------------------------------
typedef __attribute__((ext_vector_type(16))) __bf16 bf16x16;
typedef __attribute__((ext_vector_type(8)))  float  f32x8;

union Frag { bf16x16 v; uint4 q[2]; };

__device__ __forceinline__ unsigned short f2bf(float f) {
  unsigned u = __float_as_uint(f);
  u += 0x7FFFu + ((u >> 16) & 1u);     // round-to-nearest-even
  return (unsigned short)(u >> 16);
}
__device__ __forceinline__ unsigned pack2bf(float a, float b) {
  return (unsigned)f2bf(a) | ((unsigned)f2bf(b) << 16);
}

// float atomic max via sign-split integer atomics (valid with -inf init)
__device__ __forceinline__ void atomicMaxF(float* addr, float val) {
  if (val >= 0.0f) atomicMax((int*)addr, __float_as_int(val));
  else             atomicMin((unsigned int*)addr, __float_as_uint(val));
}

// ---------------------------------------------------------------------------
// Weight prep: f32 (K x Nsrc) -> bf16 pair-packed (Kpad/2 x Npad) + padded bias
// ---------------------------------------------------------------------------
__global__ void pack_weight_bias(const float* __restrict__ W, const float* __restrict__ b,
                                 unsigned* __restrict__ Wp, float* __restrict__ bp,
                                 int K, int Nsrc, int Kpad, int Npad) {
  int idx = blockIdx.x * blockDim.x + threadIdx.x;
  int total = (Kpad >> 1) * Npad;
  if (idx < total) {
    int p = idx / Npad, n = idx % Npad;
    int k0 = 2 * p;
    float a = (k0     < K && n < Nsrc) ? W[k0 * Nsrc + n]       : 0.0f;
    float c = (k0 + 1 < K && n < Nsrc) ? W[(k0 + 1) * Nsrc + n] : 0.0f;
    Wp[idx] = pack2bf(a, c);
  }
  if (idx < Npad) bp[idx] = (idx < Nsrc) ? b[idx] : 0.0f;
}

// BN stats -> per-column affine (s, t):  y = s*h + t
__global__ void bn_stats(const float* __restrict__ gSum, const float* __restrict__ gSS,
                         const float* __restrict__ g, const float* __restrict__ be,
                         float* __restrict__ s, float* __restrict__ t, int C, float invM) {
  int c = blockIdx.x * blockDim.x + threadIdx.x;
  if (c >= C) return;
  float mu  = gSum[c] * invM;
  float var = gSS[c] * invM - mu * mu;
  float sc  = g[c] * rsqrtf(var + BN_EPS);
  s[c] = sc;
  t[c] = be[c] - mu * sc;
}

// Fold BN affine of the PREVIOUS layer into this Linear:
// W'[k][n] = s[k]*W[k][n];  b'[n] = b[n] + sum_k t[k]*W[k][n]   (bf16 pair-packed out)
__global__ void fold_bn_weight(const float* __restrict__ W, const float* __restrict__ b,
                               const float* __restrict__ s, const float* __restrict__ t,
                               unsigned* __restrict__ Wp, float* __restrict__ bp,
                               int K, int Nsrc, int Kpad, int Npad) {
  int n = blockIdx.x * blockDim.x + threadIdx.x;
  if (n >= Npad) return;
  float acc = (n < Nsrc) ? b[n] : 0.0f;
  for (int p = 0; p < (Kpad >> 1); ++p) {
    int k0 = 2 * p;
    float w0 = 0.0f, w1 = 0.0f;
    if (n < Nsrc) {
      if (k0 < K)     { float w = W[k0 * Nsrc + n];       w0 = s[k0] * w;     acc += t[k0] * w; }
      if (k0 + 1 < K) { float w = W[(k0 + 1) * Nsrc + n]; w1 = s[k0 + 1] * w; acc += t[k0 + 1] * w; }
    }
    Wp[p * Npad + n] = pack2bf(w0, w1);
  }
  bp[n] = acc;
}

// ---------------------------------------------------------------------------
// Main GEMM: H = ReLU(A @ W + b), plus per-column sum / sum-of-squares.
// MODE 0: A rows are edge features [x_dst, x_src - x_dst] gathered on the fly.
// MODE 1: A is a dense f32 matrix (row stride lda, even).  In-place
//         (Hout==A) is safe: rows are block-disjoint, reads precede stores.
// Block: 256 threads (8 waves). Tile: 128 rows x 64 cols. K tiled by 32 (bf16).
// LDS layouts give each lane CONTIGUOUS 16B-aligned fragment reads:
//   As[row][20]   : 16 bf16-pairs/row; A frag = pairs [half*4..+3],[8+half*4..+3]
//   WsT[col][132] : pair-major per column; B frag = pairs [p0..p0+3],[p0+8..p0+11]
// ---------------------------------------------------------------------------
#define TM 128
template <int MODE>
__global__ __launch_bounds__(256) void gemm_wmma(
    const float* __restrict__ A, int lda, int dfeat,
    const int* __restrict__ srcI, const int* __restrict__ dstI,
    int M, int Kpad,
    const unsigned* __restrict__ Wp, const float* __restrict__ bp, int Npad,
    float* __restrict__ Hout, int ldo,
    float* __restrict__ gSum, float* __restrict__ gSS) {

  __shared__ unsigned As[TM][20];      // 128 rows x 16 pairs (+pad to 16B mult)
  __shared__ unsigned WsT[64][132];    // 64 cols x up to 128 pairs (+pad)
  __shared__ float ldsSum[64], ldsSS[64];

  const int tid  = threadIdx.x;
  const int wave = tid >> 5;
  const int lane = tid & 31;
  const int half = lane >> 4;
  const int ln   = lane & 15;
  const int blockRow = blockIdx.x * TM;
  const int cg = blockIdx.y;           // 64-column group

  if (tid < 64) { ldsSum[tid] = 0.0f; ldsSS[tid] = 0.0f; }

  // Stage weight panel column-major (pair index contiguous per column)
  {
    int pairs = (Kpad >> 1) * 64;
    for (int idx = tid; idx < pairs; idx += 256) {
      int p = idx >> 6, c = idx & 63;
      WsT[c][p] = Wp[p * Npad + cg * 64 + c];
    }
  }
  __syncthreads();

  f32x8 acc[4] = {};
  const int twoD = 2 * dfeat;

  for (int kk = 0; kk < Kpad; kk += 32) {
    __syncthreads();
    // Stage A tile: 128 rows x 32 k (16 bf16-pairs per row)
    #pragma unroll
    for (int it = 0; it < 8; ++it) {
      int idx = it * 256 + tid;        // 0..2047
      int r = idx >> 4, pc = idx & 15;
      int row = blockRow + r;
      int k0 = kk + pc * 2;
      float a0 = 0.0f, a1 = 0.0f;
      if (row < M) {
        if (MODE == 1) {
          float2 ab = *(const float2*)&A[(long long)row * lda + k0];
          a0 = ab.x; a1 = ab.y;
        } else {
          // branch-free gather of e = [x_dst, x_src - x_dst] (zero-pad k>=2d)
          int si = srcI[row], di = dstI[row];
          int c0 = (k0 < twoD) ? ((k0 < dfeat) ? k0 : k0 - dfeat) : 0;
          int c1 = (k0 + 1 < twoD) ? ((k0 + 1 < dfeat) ? k0 + 1 : k0 + 1 - dfeat) : 0;
          float xi0 = A[(long long)di * lda + c0];
          float xj0 = A[(long long)si * lda + c0];
          float xi1 = A[(long long)di * lda + c1];
          float xj1 = A[(long long)si * lda + c1];
          a0 = (k0 < dfeat) ? xi0 : (xj0 - xi0);
          a1 = (k0 + 1 < dfeat) ? xi1 : (xj1 - xi1);
          a0 = (k0 < twoD) ? a0 : 0.0f;
          a1 = (k0 + 1 < twoD) ? a1 : 0.0f;
        }
      }
      As[r][pc] = pack2bf(a0, a1);
    }
    __syncthreads();

    // A fragment: two contiguous 16B LDS reads per lane
    Frag af;
    {
      const unsigned* arow = &As[wave * 16 + ln][0];
      af.q[0] = *(const uint4*)(arow + half * 4);
      af.q[1] = *(const uint4*)(arow + 8 + half * 4);
    }

    #pragma unroll
    for (int nt = 0; nt < 4; ++nt) {
      Frag bf_;
      const unsigned* bcol = &WsT[nt * 16 + ln][(kk >> 1) + half * 4];
      bf_.q[0] = *(const uint4*)(bcol);
      bf_.q[1] = *(const uint4*)(bcol + 8);
      acc[nt] = __builtin_amdgcn_wmma_f32_16x16x32_bf16(
          false, af.v, false, bf_.v, (short)0, acc[nt], false, false);
    }
  }

  // Epilogue: bias + ReLU + store + column stats
  #pragma unroll
  for (int nt = 0; nt < 4; ++nt) {
    int col = cg * 64 + nt * 16 + ln;
    float bs = bp[col];
    float psum = 0.0f, pss = 0.0f;
    #pragma unroll
    for (int i = 0; i < 8; ++i) {
      int row = blockRow + wave * 16 + half * 8 + i;
      float y = fmaxf(acc[nt][i] + bs, 0.0f);
      if (row < M) {
        Hout[(long long)row * ldo + col] = y;
        psum += y; pss += y * y;
      }
    }
    psum += __shfl_xor(psum, 16, 32);
    pss  += __shfl_xor(pss, 16, 32);
    if (half == 0) {
      atomicAdd(&ldsSum[nt * 16 + ln], psum);
      atomicAdd(&ldsSS[nt * 16 + ln], pss);
    }
  }
  __syncthreads();
  if (tid < 64) {
    atomicAdd(&gSum[cg * 64 + tid], ldsSum[tid]);
    atomicAdd(&gSS[cg * 64 + tid],  ldsSS[tid]);
  }
}

// ---------------------------------------------------------------------------
// Elementwise helpers
// ---------------------------------------------------------------------------
__global__ void fill_f32(float* __restrict__ p, float v, long long n) {
  long long i = (long long)blockIdx.x * blockDim.x + threadIdx.x;
  if (i < n) p[i] = v;
}

// Apply BN (s,t) to edge hidden and segment-max into agg[dst]
__global__ void bn_scatter_max(const float* __restrict__ H, const float* __restrict__ s,
                               const float* __restrict__ t, const int* __restrict__ dstI,
                               float* __restrict__ agg, long long total) {
  long long idx = (long long)blockIdx.x * blockDim.x + threadIdx.x;
  if (idx >= total) return;
  int c = (int)(idx & 63);
  long long e = idx >> 6;
  float y = s[c] * H[idx] + t[c];
  atomicMaxF(&agg[(long long)dstI[e] * 64 + c], y);
}

// agg (-inf for isolated nodes -> 0) into column slice of X123
__global__ void finalize_agg(const float* __restrict__ agg, float* __restrict__ X,
                             int ldx, int colOff, long long total) {
  long long idx = (long long)blockIdx.x * blockDim.x + threadIdx.x;
  if (idx >= total) return;
  int c = (int)(idx & 63);
  long long n = idx >> 6;
  float v = agg[idx];
  if (!isfinite(v)) v = 0.0f;
  X[n * ldx + colOff + c] = v;
}

// Fold BN_m2 into Wo (f32 out, 32x24)
__global__ void fold_out(const float* __restrict__ Wo, const float* __restrict__ bo,
                         const float* __restrict__ s, const float* __restrict__ t,
                         float* __restrict__ WoF, float* __restrict__ boF) {
  int c = threadIdx.x;
  if (c >= 24) return;
  float acc = bo[c];
  for (int k = 0; k < 32; ++k) {
    float w = Wo[k * 24 + c];
    WoF[k * 24 + c] = s[k] * w;
    acc += t[k] * w;
  }
  boF[c] = acc;
}

// Final 32->24 linear + log_softmax (per node)
__global__ void final_head(const float* __restrict__ H, int ldh,
                           const float* __restrict__ WoF, const float* __restrict__ boF,
                           float* __restrict__ out, int N) {
  int n = blockIdx.x * blockDim.x + threadIdx.x;
  if (n >= N) return;
  float h[32];
  #pragma unroll
  for (int k = 0; k < 32; ++k) h[k] = H[(long long)n * ldh + k];
  float logits[24];
  float m = -1e30f;
  for (int c = 0; c < 24; ++c) {
    float a = boF[c];
    #pragma unroll
    for (int k = 0; k < 32; ++k) a += h[k] * WoF[k * 24 + c];
    logits[c] = a;
    m = fmaxf(m, a);
  }
  float sum = 0.0f;
  for (int c = 0; c < 24; ++c) sum += __expf(logits[c] - m);
  float lse = m + __logf(sum);
  for (int c = 0; c < 24; ++c) out[(long long)n * 24 + c] = logits[c] - lse;
}

// ---------------------------------------------------------------------------
// Host orchestration
// ---------------------------------------------------------------------------
extern "C" void kernel_launch(void* const* d_in, const int* in_sizes, int n_in,
                              void* d_out, int out_size, void* d_ws, size_t ws_size,
                              hipStream_t stream) {
  const int N = in_sizes[0] / 6;
  const int E = in_sizes[31] / 2;

  const float* x    = (const float*)d_in[0];
  const float* W1a  = (const float*)d_in[1];  const float* b1a  = (const float*)d_in[2];
  const float* g1a  = (const float*)d_in[3];  const float* be1a = (const float*)d_in[4];
  const float* W1b  = (const float*)d_in[5];  const float* b1b  = (const float*)d_in[6];
  const float* g1b  = (const float*)d_in[7];  const float* be1b = (const float*)d_in[8];
  const float* Wa   = (const float*)d_in[9];  const float* ba   = (const float*)d_in[10];
  const float* ga   = (const float*)d_in[11]; const float* bea  = (const float*)d_in[12];
  const float* Wb   = (const float*)d_in[13]; const float* bb   = (const float*)d_in[14];
  const float* gb   = (const float*)d_in[15]; const float* beb  = (const float*)d_in[16];
  const float* Wl   = (const float*)d_in[17]; const float* bl   = (const float*)d_in[18];
  const float* gl   = (const float*)d_in[19]; const float* bel  = (const float*)d_in[20];
  const float* Wm1  = (const float*)d_in[21]; const float* bm1  = (const float*)d_in[22];
  const float* gm1  = (const float*)d_in[23]; const float* bem1 = (const float*)d_in[24];
  const float* Wm2  = (const float*)d_in[25]; const float* bm2  = (const float*)d_in[26];
  const float* gm2  = (const float*)d_in[27]; const float* bem2 = (const float*)d_in[28];
  const float* Wo   = (const float*)d_in[29]; const float* bo   = (const float*)d_in[30];
  const int*   ei   = (const int*)d_in[31];
  const int* srcI = ei;
  const int* dstI = ei + E;
  float* out = (float*)d_out;

  // ---- carve workspace ----
  char* wsp = (char*)d_ws;
  auto carve = [&](size_t bytes) -> void* {
    void* p = (void*)wsp;
    wsp += (bytes + 255) & ~(size_t)255;
    return p;
  };
  float* hEdge = (float*)carve((size_t)E * 64 * 4);   // edge hidden (reused/in-place)
  float* X123  = (float*)carve((size_t)N * 192 * 4);  // [x1|x2|x3]
  float* hL    = (float*)carve((size_t)N * 256 * 4);
  float* hM1   = (float*)carve((size_t)N * 64 * 4);
  float* hM2   = (float*)carve((size_t)N * 64 * 4);   // cols 32..63 are zero pad
  float* agg   = (float*)carve((size_t)N * 64 * 4);
  float* gSum  = (float*)carve(512 * 4);              // gSS = gSum+256
  float* gSS   = gSum + 256;
  float* sArr  = (float*)carve(256 * 4);
  float* tArr  = (float*)carve(256 * 4);
  unsigned* W1aP = (unsigned*)carve(16 * 64 * 4);     // Kpad=32
  unsigned* W1bP = (unsigned*)carve(32 * 64 * 4);     // Kpad=64
  unsigned* Wa0P = (unsigned*)carve(64 * 64 * 4);     // Kpad=128
  unsigned* Wa1P = (unsigned*)carve(64 * 64 * 4);
  unsigned* Wb0P = (unsigned*)carve(32 * 64 * 4);
  unsigned* Wb1P = (unsigned*)carve(32 * 64 * 4);
  unsigned* WlP  = (unsigned*)carve(96 * 256 * 4);    // Kpad=192, Npad=256
  unsigned* Wm1P = (unsigned*)carve(128 * 64 * 4);    // Kpad=256
  unsigned* Wm2P = (unsigned*)carve(32 * 64 * 4);     // Kpad=64, Npad=64 (Nsrc=32)
  float* b1aP = (float*)carve(64 * 4);
  float* b1bP = (float*)carve(64 * 4);
  float* ba0P = (float*)carve(64 * 4);
  float* ba1P = (float*)carve(64 * 4);
  float* bb0P = (float*)carve(64 * 4);
  float* bb1P = (float*)carve(64 * 4);
  float* blP  = (float*)carve(256 * 4);
  float* bm1P = (float*)carve(64 * 4);
  float* bm2P = (float*)carve(64 * 4);
  float* WoF  = (float*)carve(32 * 24 * 4);
  float* boF  = (float*)carve(24 * 4);
  (void)ws_size; (void)n_in; (void)out_size;

  const float invE = 1.0f / (float)E;
  const float invN = 1.0f / (float)N;
  const long long E64 = (long long)E * 64;
  const long long N64 = (long long)N * 64;
  const dim3 gE((E + TM - 1) / TM, 1);
  const dim3 gN((N + TM - 1) / TM, 1);

  // ---- pack first-linear weights (no fold needed) ----
  pack_weight_bias<<<(16 * 64 + 255) / 256, 256, 0, stream>>>(W1a, b1a, W1aP, b1aP, 12, 64, 32, 64);
  pack_weight_bias<<<(64 * 64 + 255) / 256, 256, 0, stream>>>(Wa,            ba,      Wa0P, ba0P, 128, 64, 128, 64);
  pack_weight_bias<<<(64 * 64 + 255) / 256, 256, 0, stream>>>(Wa + 128 * 64, ba + 64, Wa1P, ba1P, 128, 64, 128, 64);
  pack_weight_bias<<<(96 * 256 + 255) / 256, 256, 0, stream>>>(Wl, bl, WlP, blP, 192, 256, 192, 256);

  // ---- one EdgeConv ----
  auto run_conv = [&](const float* xin, int lda, int dfeat, int Kpad1,
                      const unsigned* WAp, const float* bAp,
                      const float* Wb_raw, const float* bb_raw,
                      const float* gA, const float* beA,
                      const float* gB, const float* beB,
                      unsigned* WBp, float* bBp, int slice) {
    // Lin-a + ReLU (+ stats of h_a)
    hipMemsetAsync(gSum, 0, 512 * sizeof(float), stream);
    gemm_wmma<0><<<gE, 256, 0, stream>>>(xin, lda, dfeat, srcI, dstI, E, Kpad1,
                                         WAp, bAp, 64, hEdge, 64, gSum, gSS);
    bn_stats<<<1, 64, 0, stream>>>(gSum, gSS, gA, beA, sArr, tArr, 64, invE);
    // fold BN-a into Lin-b
    fold_bn_weight<<<1, 64, 0, stream>>>(Wb_raw, bb_raw, sArr, tArr, WBp, bBp, 64, 64, 64, 64);
    // Lin-b + ReLU (in-place) (+ stats of h_b)
    hipMemsetAsync(gSum, 0, 512 * sizeof(float), stream);
    gemm_wmma<1><<<gE, 256, 0, stream>>>(hEdge, 64, 0, nullptr, nullptr, E, 64,
                                         WBp, bBp, 64, hEdge, 64, gSum, gSS);
    bn_stats<<<1, 64, 0, stream>>>(gSum, gSS, gB, beB, sArr, tArr, 64, invE);
    // BN-b + segment max -> X123 slice
    fill_f32<<<(unsigned)((N64 + 255) / 256), 256, 0, stream>>>(agg, -__builtin_huge_valf(), N64);
    bn_scatter_max<<<(unsigned)((E64 + 255) / 256), 256, 0, stream>>>(hEdge, sArr, tArr, dstI, agg, E64);
    finalize_agg<<<(unsigned)((N64 + 255) / 256), 256, 0, stream>>>(agg, X123, 192, slice * 64, N64);
  };

  run_conv(x,         6,   6,  32,  W1aP, b1aP, W1b, b1b, g1a, be1a, g1b, be1b, W1bP, b1bP, 0);
  run_conv(X123,      192, 64, 128, Wa0P, ba0P, Wb,           bb,      ga,      bea,      gb,      beb,      Wb0P, bb0P, 1);
  run_conv(X123 + 64, 192, 64, 128, Wa1P, ba1P, Wb + 64 * 64, bb + 64, ga + 64, bea + 64, gb + 64, beb + 64, Wb1P, bb1P, 2);

  // ---- lin1: (N x 192) @ (192 x 256) + ReLU (+ stats) ----
  hipMemsetAsync(gSum, 0, 512 * sizeof(float), stream);
  gemm_wmma<1><<<dim3((N + TM - 1) / TM, 4), 256, 0, stream>>>(X123, 192, 0, nullptr, nullptr,
                                                               N, 192, WlP, blP, 256, hL, 256, gSum, gSS);
  bn_stats<<<1, 256, 0, stream>>>(gSum, gSS, gl, bel, sArr, tArr, 256, invN);

  // ---- m1: fold BN_l, (N x 256) @ (256 x 64) + ReLU (+ stats) ----
  fold_bn_weight<<<1, 64, 0, stream>>>(Wm1, bm1, sArr, tArr, Wm1P, bm1P, 256, 64, 256, 64);
  hipMemsetAsync(gSum, 0, 512 * sizeof(float), stream);
  gemm_wmma<1><<<gN, 256, 0, stream>>>(hL, 256, 0, nullptr, nullptr, N, 256,
                                       Wm1P, bm1P, 64, hM1, 64, gSum, gSS);
  bn_stats<<<1, 64, 0, stream>>>(gSum, gSS, gm1, bem1, sArr, tArr, 64, invN);

  // ---- m2: fold BN_m1, (N x 64) @ (64 x 32pad64) + ReLU (+ stats cols 0..31) ----
  fold_bn_weight<<<1, 64, 0, stream>>>(Wm2, bm2, sArr, tArr, Wm2P, bm2P, 64, 32, 64, 64);
  hipMemsetAsync(gSum, 0, 512 * sizeof(float), stream);
  gemm_wmma<1><<<gN, 256, 0, stream>>>(hM1, 64, 0, nullptr, nullptr, N, 64,
                                       Wm2P, bm2P, 64, hM2, 64, gSum, gSS);
  bn_stats<<<1, 64, 0, stream>>>(gSum, gSS, gm2, bem2, sArr, tArr, 32, invN);

  // ---- head: fold BN_m2 into Wo, then 32->24 linear + log_softmax ----
  fold_out<<<1, 32, 0, stream>>>(Wo, bo, sArr, tArr, WoF, boF);
  final_head<<<(N + 127) / 128, 128, 0, stream>>>(hM2, 64, WoF, boF, out, N);
}
